// GQAAttention_38628935860426
// MI455X (gfx1250) — matrix operand order repaired and checked by reading, not compiled, not run on hardware
//
#include <hip/hip_runtime.h>
#include <hip/hip_bf16.h>
#include <cstdint>

typedef __attribute__((ext_vector_type(16))) __bf16 v16bf;
typedef __attribute__((ext_vector_type(8)))  float  v8f;

static constexpr int Bc = 2, Tc = 2048, Cc = 2048, NQc = 16, NKVc = 4, Hc = 128;
static constexpr int Mrows = Bc * Tc; // 4096
static constexpr float EPSc = 1e-6f;
static constexpr float SCALEc = 0.0883883476483184f; // 1/sqrt(128)

#if __has_builtin(__builtin_amdgcn_global_load_async_to_lds_b128)
#define ASYNC_LDS 1
#else
#define ASYNC_LDS 0
#endif

__device__ __forceinline__ unsigned short f32_to_bf16(float f) {
  unsigned int u = __float_as_uint(f);
  u += 0x7fffu + ((u >> 16) & 1u); // round-to-nearest-even
  return (unsigned short)(u >> 16);
}
__device__ __forceinline__ float bf16_to_f32(unsigned short h) {
  return __uint_as_float(((unsigned int)h) << 16);
}

#if ASYNC_LDS
typedef int v4i_vec __attribute__((vector_size(16)));
typedef __attribute__((address_space(1))) v4i_vec* gptr_v4i; // global int4*
typedef __attribute__((address_space(3))) v4i_vec* lptr_v4i; // LDS int4*
#endif

// 16-byte global -> LDS copy: async (ASYNCcnt) on CDNA5, sync fallback otherwise.
__device__ __forceinline__ void async_copy_b128(const unsigned short* g, unsigned short* l) {
#if ASYNC_LDS
  __builtin_amdgcn_global_load_async_to_lds_b128((gptr_v4i)g, (lptr_v4i)l, 0, 0);
#else
  const uint4 d = *(const uint4*)g;
  unsigned int* dst = (unsigned int*)l;
  dst[0] = d.x; dst[1] = d.y; dst[2] = d.z; dst[3] = d.w;
#endif
}
__device__ __forceinline__ void wait_async() {
#if ASYNC_LDS
#if __has_builtin(__builtin_amdgcn_s_wait_asynccnt)
  __builtin_amdgcn_s_wait_asynccnt(0);
#else
  asm volatile("s_wait_asynccnt 0x0" ::: "memory");
#endif
#endif
}

union FragB { v16bf v; unsigned int u[8]; };

// ---------------------------------------------------------------- convert
__global__ __launch_bounds__(256) void cvt_f32_bf16(const float* __restrict__ in,
                                                    unsigned short* __restrict__ out,
                                                    int n) {
  for (int i = blockIdx.x * 256 + threadIdx.x; i < n; i += gridDim.x * 256)
    out[i] = f32_to_bf16(in[i]);
}

// ---------------------------------------------------------------- GEMM
// C(MxN) = A(MxK, bf16 row-major) * B(KxN, bf16 row-major)
// 128x128 block tile, K-step 32, 8 waves; wave tile 32x64 (2x4 of 16x16 WMMA).
// Double-buffered LDS; A staged with async global->LDS, B pipelined via regs.
template <bool OUT_F32>
__global__ __launch_bounds__(256) void gemm_bf16(const unsigned short* __restrict__ A,
                                                 const unsigned short* __restrict__ B,
                                                 void* __restrict__ Cp,
                                                 int M, int N, int K) {
  __shared__ unsigned short As[2][128][48]; // [m][k], row stride 96B (16B aligned)
  __shared__ unsigned short Bs[2][128][34]; // transposed: [n][k]
  const int tid = threadIdx.x, lane = tid & 31, wid = tid >> 5;
  const int l16 = lane & 15, lhalf = lane >> 4;
  const int bm = blockIdx.y * 128, bn = blockIdx.x * 128;
  const int wm = (wid >> 1) * 32, wn = (wid & 1) * 64;
  const int nk = K / 32;

  v8f acc[2][4];
#pragma unroll
  for (int i = 0; i < 2; ++i)
#pragma unroll
    for (int j = 0; j < 4; ++j)
#pragma unroll
      for (int e = 0; e < 8; ++e) acc[i][j][e] = 0.0f;

  uint4 breg[2];

  auto stageA = [&](int kt, int buf) {
#pragma unroll
    for (int p = 0; p < 2; ++p) {
      int idx = p * 256 + tid;     // 0..511 -> 128 rows x 4 chunks
      int row = idx >> 2, c4 = idx & 3;
      async_copy_b128(A + (size_t)(bm + row) * K + kt * 32 + c4 * 8,
                      &As[buf][row][c4 * 8]);
    }
  };
  auto loadB = [&](int kt) {
#pragma unroll
    for (int p = 0; p < 2; ++p) {
      int idx = p * 256 + tid;     // 0..511 -> 32 k-rows x 16 chunks
      int kr = idx >> 4, c4 = idx & 15;
      breg[p] = *(const uint4*)(B + (size_t)(kt * 32 + kr) * N + bn + c4 * 8);
    }
  };
  auto storeB = [&](int buf) {
#pragma unroll
    for (int p = 0; p < 2; ++p) {
      int idx = p * 256 + tid;
      int kr = idx >> 4, c4 = idx & 15;
      unsigned int w[4] = {breg[p].x, breg[p].y, breg[p].z, breg[p].w};
      int n0 = c4 * 8;
#pragma unroll
      for (int q = 0; q < 4; ++q) {
        Bs[buf][n0 + 2 * q][kr]     = (unsigned short)(w[q] & 0xffffu);
        Bs[buf][n0 + 2 * q + 1][kr] = (unsigned short)(w[q] >> 16);
      }
    }
  };

  // preamble: stage tile 0
  stageA(0, 0);
  loadB(0);
  storeB(0);

  for (int kt = 0; kt < nk; ++kt) {
    const int cur = kt & 1, nxt = cur ^ 1;
    wait_async();        // this wave's async A(cur) landed
    __syncthreads();     // everyone's A(cur) + B(cur) stores visible; buffers free
    if (kt + 1 < nk) {
      stageA(kt + 1, nxt);   // async copies overlap with compute below
      loadB(kt + 1);         // global loads overlap with compute below
    }

    // A fragments: ISA 16-bit A 16x32 layout (lane half -> K-group of 8)
    FragB af[2];
#pragma unroll
    for (int i = 0; i < 2; ++i)
#pragma unroll
      for (int r = 0; r < 8; ++r) {
        int kk = ((r < 4) ? 0 : 16) + lhalf * 8 + ((r & 3) << 1);
        af[i].u[r] = *(const unsigned int*)&As[cur][wm + i * 16 + l16][kk];
      }
#pragma unroll
    for (int j = 0; j < 4; ++j) {
      FragB bf;
#pragma unroll
      for (int r = 0; r < 8; ++r) {
        int kk = ((r < 4) ? 0 : 16) + lhalf * 8 + ((r & 3) << 1);
        bf.u[r] = *(const unsigned int*)&Bs[cur][wn + j * 16 + l16][kk];
      }
#pragma unroll
      for (int i = 0; i < 2; ++i)
        acc[i][j] = __builtin_amdgcn_wmma_f32_16x16x32_bf16(
            false, af[i].v, false, bf.v, (short)0, acc[i][j], false, false);
    }

    if (kt + 1 < nk) storeB(nxt);  // regs -> LDS for next tile (other buffer)
  }

  // epilogue: D layout lane(n)=l16, row m = r + 8*lhalf
#pragma unroll
  for (int i = 0; i < 2; ++i)
#pragma unroll
    for (int j = 0; j < 4; ++j)
#pragma unroll
      for (int r = 0; r < 8; ++r) {
        int row = bm + wm + i * 16 + r + 8 * lhalf;
        int col = bn + wn + j * 16 + l16;
        float v = acc[i][j][r];
        if (OUT_F32) ((float*)Cp)[(size_t)row * N + col] = v;
        else ((unsigned short*)Cp)[(size_t)row * N + col] = f32_to_bf16(v);
      }
}

// ---------------------------------------------------------------- RMSNorm + RoPE (in place, bf16)
__global__ __launch_bounds__(128) void rmsnorm_rope(unsigned short* __restrict__ buf,
                                                    const float* __restrict__ w,
                                                    const float* __restrict__ cosb,
                                                    const float* __restrict__ sinb,
                                                    int nheads) {
  __shared__ float red[128];
  const int bh = blockIdx.x;
  const int head = bh % nheads;
  const int row = bh / nheads;  // b*T + t
  const int t = row % Tc;
  const int d = threadIdx.x;
  unsigned short* p = buf + (size_t)row * nheads * Hc + head * Hc;
  float x = bf16_to_f32(p[d]);
  red[d] = x * x;
  __syncthreads();
#pragma unroll
  for (int s = 64; s > 0; s >>= 1) {
    if (d < s) red[d] += red[d + s];
    __syncthreads();
  }
  float inv = rsqrtf(red[0] * (1.0f / Hc) + EPSc);
  __syncthreads();
  float xn = x * inv * w[d];
  red[d] = xn;
  __syncthreads();
  float partner = red[(d + 64) & 127];
  float rot = (d < 64) ? -partner : partner;
  float o = xn * cosb[t * Hc + d] + rot * sinb[t * Hc + d];
  p[d] = f32_to_bf16(o);
}

// ---------------------------------------------------------------- flash attention (causal, GQA 4:1)
// grid (T/64, NQ, B); block = 4 waves, each wave owns 16 q rows.
__global__ __launch_bounds__(128) void flash_attn(const unsigned short* __restrict__ qb,
                                                  const unsigned short* __restrict__ kb,
                                                  const unsigned short* __restrict__ vb,
                                                  unsigned short* __restrict__ ctx) {
  __shared__ unsigned short Ks[32][136];   // [kv][h], row stride 272B (16B aligned)
  __shared__ unsigned short Vt[128][34];   // [h][kv]  -> B-frag for PV
  __shared__ unsigned short Ps[4][16][34]; // per-wave P tile (A layout source)
  const int tid = threadIdx.x, lane = tid & 31, wid = tid >> 5;
  const int l16 = lane & 15, lhalf = lane >> 4;
  const int b = blockIdx.z, head = blockIdx.y, kvh = head >> 2;
  const int q0blk = blockIdx.x * 64;
  const int q0 = q0blk + wid * 16;
  const size_t qrl = (size_t)NQc * Hc;   // 2048
  const size_t krl = (size_t)NKVc * Hc;  // 512

  // Q fragments (16 rows x H=128) held in registers: 4 K-chunks of 32
  FragB aq[4];
#pragma unroll
  for (int c = 0; c < 4; ++c)
#pragma unroll
    for (int r = 0; r < 8; ++r) {
      int hh = c * 32 + ((r < 4) ? 0 : 16) + lhalf * 8 + ((r & 3) << 1);
      aq[c].u[r] = *(const unsigned int*)(qb + (size_t)(b * Tc + q0 + l16) * qrl + head * Hc + hh);
    }

  v8f O[8];
#pragma unroll
  for (int nb = 0; nb < 8; ++nb)
#pragma unroll
    for (int e = 0; e < 8; ++e) O[nb][e] = 0.0f;
  float mrow[8], lrow[8];
#pragma unroll
  for (int v = 0; v < 8; ++v) { mrow[v] = -1e30f; lrow[v] = 0.0f; }

  const int ntiles = (q0blk + 64) / 32;
  for (int jt = 0; jt < ntiles; ++jt) {
    const int j0 = jt * 32;
    __syncthreads();
    // K tile via async global->LDS; V tile transposed through VGPRs
#pragma unroll
    for (int p = 0; p < 4; ++p) {
      int idx = p * 128 + tid;  // 0..511 -> 32 kv-rows x 16 chunks
      int kr = idx >> 4, c4 = idx & 15;
      async_copy_b128(kb + (size_t)(b * Tc + j0 + kr) * krl + kvh * Hc + c4 * 8,
                      &Ks[kr][c4 * 8]);
      const uint4 d = *(const uint4*)(vb + (size_t)(b * Tc + j0 + kr) * krl + kvh * Hc + c4 * 8);
      unsigned int w[4] = {d.x, d.y, d.z, d.w};
      int h0 = c4 * 8;
#pragma unroll
      for (int q = 0; q < 4; ++q) {
        Vt[h0 + 2 * q][kr]     = (unsigned short)(w[q] & 0xffffu);
        Vt[h0 + 2 * q + 1][kr] = (unsigned short)(w[q] >> 16);
      }
    }
    wait_async();
    __syncthreads();

    // S = Q K^T  (two 16x16 blocks: kv 0-15, kv 16-31)
    v8f s0, s1;
#pragma unroll
    for (int e = 0; e < 8; ++e) { s0[e] = 0.0f; s1[e] = 0.0f; }
#pragma unroll
    for (int c = 0; c < 4; ++c) {
      FragB bk0, bk1;
#pragma unroll
      for (int r = 0; r < 8; ++r) {
        int kk = c * 32 + ((r < 4) ? 0 : 16) + lhalf * 8 + ((r & 3) << 1);
        bk0.u[r] = *(const unsigned int*)&Ks[l16][kk];
        bk1.u[r] = *(const unsigned int*)&Ks[16 + l16][kk];
      }
      s0 = __builtin_amdgcn_wmma_f32_16x16x32_bf16(false, aq[c].v, false, bk0.v, (short)0, s0, false, false);
      s1 = __builtin_amdgcn_wmma_f32_16x16x32_bf16(false, aq[c].v, false, bk1.v, (short)0, s1, false, false);
    }

    // online softmax: row m = v + 8*lhalf, cols spread over 16 lanes (wave32 xor-shuffles)
#pragma unroll
    for (int v = 0; v < 8; ++v) {
      int m = v + 8 * lhalf;
      int qi = q0 + m;
      float a0 = s0[v] * SCALEc + ((j0 + l16)      <= qi ? 0.0f : -1e9f);
      float a1 = s1[v] * SCALEc + ((j0 + 16 + l16) <= qi ? 0.0f : -1e9f);
      float mx = fmaxf(a0, a1);
#pragma unroll
      for (int off = 1; off < 16; off <<= 1)
        mx = fmaxf(mx, __shfl_xor(mx, off, 32));
      float mn = fmaxf(mrow[v], mx);
      float alpha = __expf(mrow[v] - mn);
      float e0 = __expf(a0 - mn), e1 = __expf(a1 - mn);
      float rs = e0 + e1;
#pragma unroll
      for (int off = 1; off < 16; off <<= 1)
        rs += __shfl_xor(rs, off, 32);
      lrow[v] = lrow[v] * alpha + rs;
      mrow[v] = mn;
#pragma unroll
      for (int nb = 0; nb < 8; ++nb) O[nb][v] *= alpha;
      Ps[wid][m][l16]      = f32_to_bf16(e0);
      Ps[wid][m][16 + l16] = f32_to_bf16(e1);
    }

    // O += P * V
    FragB ap;
#pragma unroll
    for (int r = 0; r < 8; ++r) {
      int kk = ((r < 4) ? 0 : 16) + lhalf * 8 + ((r & 3) << 1);
      ap.u[r] = *(const unsigned int*)&Ps[wid][l16][kk];
    }
#pragma unroll
    for (int nb = 0; nb < 8; ++nb) {
      FragB bv;
#pragma unroll
      for (int r = 0; r < 8; ++r) {
        int kk = ((r < 4) ? 0 : 16) + lhalf * 8 + ((r & 3) << 1);
        bv.u[r] = *(const unsigned int*)&Vt[nb * 16 + l16][kk];
      }
      O[nb] = __builtin_amdgcn_wmma_f32_16x16x32_bf16(false, ap.v, false, bv.v, (short)0, O[nb], false, false);
    }
  }

  // finalize: ctx[b,t,head*H + h] = O / l
#pragma unroll
  for (int nb = 0; nb < 8; ++nb)
#pragma unroll
    for (int v = 0; v < 8; ++v) {
      int m = v + 8 * lhalf;
      size_t row = (size_t)(b * Tc + q0 + m);
      int col = head * Hc + nb * 16 + l16;
      ctx[row * qrl + col] = f32_to_bf16(O[nb][v] / lrow[v]);
    }
}

// ---------------------------------------------------------------- launch
extern "C" void kernel_launch(void* const* d_in, const int* in_sizes, int n_in,
                              void* d_out, int out_size, void* d_ws, size_t ws_size,
                              hipStream_t stream) {
  (void)in_sizes; (void)n_in; (void)out_size; (void)ws_size;
  const float* x    = (const float*)d_in[0];
  const float* Wq   = (const float*)d_in[1];
  const float* Wk   = (const float*)d_in[2];
  const float* Wv   = (const float*)d_in[3];
  const float* Wo   = (const float*)d_in[4];
  const float* qw   = (const float*)d_in[5];
  const float* kw   = (const float*)d_in[6];
  const float* cosb = (const float*)d_in[7];
  const float* sinb = (const float*)d_in[8];
  // d_in[9] (attn_mask) unused: causal mask computed inline.

  char* ws = (char*)d_ws;
  size_t off = 0;
  auto alloc = [&](size_t bytes) {
    char* p = ws + off;
    off += (bytes + 255) & ~(size_t)255;
    return p;
  };
  unsigned short* xb   = (unsigned short*)alloc((size_t)Mrows * Cc * 2);
  unsigned short* Wqb  = (unsigned short*)alloc((size_t)Cc * NQc * Hc * 2);
  unsigned short* Wkb  = (unsigned short*)alloc((size_t)Cc * NKVc * Hc * 2);
  unsigned short* Wvb  = (unsigned short*)alloc((size_t)Cc * NKVc * Hc * 2);
  unsigned short* Wob  = (unsigned short*)alloc((size_t)NQc * Hc * Cc * 2);
  unsigned short* qbuf = (unsigned short*)alloc((size_t)Mrows * NQc * Hc * 2);
  unsigned short* kbuf = (unsigned short*)alloc((size_t)Mrows * NKVc * Hc * 2);
  unsigned short* vbuf = (unsigned short*)alloc((size_t)Mrows * NKVc * Hc * 2);
  unsigned short* ctxb = (unsigned short*)alloc((size_t)Mrows * NQc * Hc * 2);

  auto cvt = [&](const float* src, unsigned short* dst, int n) {
    cvt_f32_bf16<<<dim3((n + 255) / 256), dim3(256), 0, stream>>>(src, dst, n);
  };
  cvt(x,  xb,  Mrows * Cc);
  cvt(Wq, Wqb, Cc * NQc * Hc);
  cvt(Wk, Wkb, Cc * NKVc * Hc);
  cvt(Wv, Wvb, Cc * NKVc * Hc);
  cvt(Wo, Wob, NQc * Hc * Cc);

  // QKV projections (bf16 WMMA, bf16 out)
  gemm_bf16<false><<<dim3(16, 32), dim3(256), 0, stream>>>(xb, Wqb, qbuf, Mrows, 2048, 2048);
  gemm_bf16<false><<<dim3(4, 32),  dim3(256), 0, stream>>>(xb, Wkb, kbuf, Mrows, 512, 2048);
  gemm_bf16<false><<<dim3(4, 32),  dim3(256), 0, stream>>>(xb, Wvb, vbuf, Mrows, 512, 2048);

  // RMSNorm + RoPE (in place)
  rmsnorm_rope<<<dim3(Mrows * NQc),  dim3(128), 0, stream>>>(qbuf, qw, cosb, sinb, NQc);
  rmsnorm_rope<<<dim3(Mrows * NKVc), dim3(128), 0, stream>>>(kbuf, kw, cosb, sinb, NKVc);

  // causal GQA flash attention
  flash_attn<<<dim3(Tc / 64, NQc, Bc), dim3(128), 0, stream>>>(qbuf, kbuf, vbuf, ctxb);

  // output projection (f32 out)
  gemm_bf16<true><<<dim3(16, 32), dim3(256), 0, stream>>>(ctxb, Wob, d_out, Mrows, 2048, 2048);
}
